// MultiheadSelfAttentionDec_52785148067957
// MI455X (gfx1250) — matrix operand order repaired and checked by reading, not compiled, not run on hardware
//
#include <hip/hip_runtime.h>

// ---------------- problem constants ----------------
#define Sx 2048
#define Cx 3072
#define Hx 24
#define HDx 128
#define EPSx 1e-6f

// ---------------- vector types ----------------
typedef __attribute__((ext_vector_type(16))) __bf16 bf16x16;
typedef __attribute__((ext_vector_type(8)))  __bf16 bf16x8;
typedef __attribute__((ext_vector_type(8)))  float  f32x8;

__device__ __forceinline__ bf16x16 frag_from(const __bf16* p0, const __bf16* p1) {
    union { bf16x16 v; bf16x8 h[2]; } u;
    u.h[0] = *(const bf16x8*)p0;
    u.h[1] = *(const bf16x8*)p1;
    return u.v;
}

__device__ __forceinline__ f32x8 wmma_bf16(bf16x16 a, bf16x16 b, f32x8 c) {
    // 8 args: (neg_a, A, neg_b, B, c_mod, C, reuse_a, reuse_b)
    return __builtin_amdgcn_wmma_f32_16x16x32_bf16(false, a, false, b, (short)0, c, false, false);
}

// ---------------- CDNA5 async global->LDS (ASYNCcnt-tracked) ----------------
// GV-mode VGLOBAL encoding: global_load_async_to_lds_b128 vdst(lds), v[addr], off
// Generic shared pointers carry the LDS byte offset in their low 32 bits
// (ISA aperture rule), so truncation is the LDS address.
__device__ __forceinline__ void async_load_b128(const __bf16* gptr, __bf16* lptr) {
    unsigned laddr = (unsigned)(unsigned long long)lptr;
    asm volatile("global_load_async_to_lds_b128 %0, %1, off"
                 :: "v"(laddr), "v"((unsigned long long)gptr) : "memory");
}

#if __has_builtin(__builtin_amdgcn_s_wait_asynccnt)
#define WAIT_ASYNCCNT(n) __builtin_amdgcn_s_wait_asynccnt(n)
#else
#define WAIT_ASYNCCNT(n) asm volatile("s_wait_asynccnt %0" :: "i"(n) : "memory")
#endif

// ---------------- fp32 -> bf16 conversion ----------------
__global__ __launch_bounds__(256)
void cvt_f32_bf16(const float* __restrict__ src, __bf16* __restrict__ dst, size_t n) {
    size_t i = ((size_t)blockIdx.x * 256 + threadIdx.x) * 4;
    if (i + 3 < n) {
        float4 v = *(const float4*)(src + i);
        dst[i + 0] = (__bf16)v.x;
        dst[i + 1] = (__bf16)v.y;
        dst[i + 2] = (__bf16)v.z;
        dst[i + 3] = (__bf16)v.w;
    }
}

// ---------------- 128x128 NT GEMM: out[m,n] = sum_k A[m,k] * B[n,k] + bias[n] ----------------
// A: [M,K] bf16 row-major, B: [N,K] bf16 row-major. One of outF/outB is non-null.
// Double-buffered LDS staging via async loads; compute overlaps next tile's DMA.
#define GTM 128
#define GTN 128
#define GTK 64

__global__ __launch_bounds__(256)
void gemm_nt_bias(const __bf16* __restrict__ A, const __bf16* __restrict__ B,
                  const float* __restrict__ bias,
                  float* __restrict__ outF, __bf16* __restrict__ outB,
                  int M, int N, int K) {
    __shared__ alignas(16) __bf16 As[2][GTM][GTK];   // 32 KB
    __shared__ alignas(16) __bf16 Bs[2][GTN][GTK];   // 32 KB

    const int tid  = threadIdx.x;
    const int wave = tid >> 5;
    const int lane = tid & 31;
    const int hi   = lane >> 4;
    const int lr   = lane & 15;

    const int m0 = blockIdx.x * GTM;
    const int n0 = blockIdx.y * GTN;

    // 8 waves in 4(M) x 2(N) grid; wave tile 32(M) x 64(N)
    const int wm = (wave & 3) * 32;
    const int wn = (wave >> 2) * 64;

    const f32x8 zero8 = {0.f,0.f,0.f,0.f,0.f,0.f,0.f,0.f};
    f32x8 acc[2][4];
#pragma unroll
    for (int i = 0; i < 2; ++i)
#pragma unroll
        for (int j = 0; j < 4; ++j) acc[i][j] = zero8;

    const int ldr = tid >> 1;            // 128 rows
    const int ldc = (tid & 1) * 32;      // 2 x 32 cols
    const __bf16* agbase = A + (size_t)(m0 + ldr) * K + ldc;
    const __bf16* bgbase = B + (size_t)(n0 + ldr) * K + ldc;

    auto stage = [&](int buf, int kb) {
#pragma unroll
        for (int u = 0; u < 4; ++u) {
            async_load_b128(agbase + kb + u * 8, &As[buf][ldr][ldc + u * 8]);
            async_load_b128(bgbase + kb + u * 8, &Bs[buf][ldr][ldc + u * 8]);
        }
    };

    const int T = K / GTK;
    stage(0, 0);
    for (int t = 0; t < T; ++t) {
        const int cur = t & 1;
        if (t + 1 < T) {
            stage(cur ^ 1, (t + 1) * GTK);
            WAIT_ASYNCCNT(8);            // tile t resident; tile t+1 in flight
        } else {
            WAIT_ASYNCCNT(0);
        }
        __syncthreads();

#pragma unroll
        for (int kc = 0; kc < GTK; kc += 32) {
            bf16x16 af[2], bf_[4];
#pragma unroll
            for (int i = 0; i < 2; ++i)
                af[i] = frag_from(&As[cur][wm + i * 16 + lr][kc + hi * 8],
                                  &As[cur][wm + i * 16 + lr][kc + 16 + hi * 8]);
#pragma unroll
            for (int j = 0; j < 4; ++j)
                bf_[j] = frag_from(&Bs[cur][wn + j * 16 + lr][kc + hi * 8],
                                   &Bs[cur][wn + j * 16 + lr][kc + 16 + hi * 8]);
#pragma unroll
            for (int i = 0; i < 2; ++i)
#pragma unroll
                for (int j = 0; j < 4; ++j)
                    acc[i][j] = wmma_bf16(af[i], bf_[j], acc[i][j]);
        }
        __syncthreads();
    }

    // epilogue: element r of lane -> (M = r + 8*hi, N = lr)
#pragma unroll
    for (int j = 0; j < 4; ++j) {
        const int n  = n0 + wn + j * 16 + lr;
        const float bv = bias ? bias[n] : 0.f;
#pragma unroll
        for (int i = 0; i < 2; ++i) {
            const int mb = m0 + wm + i * 16 + hi * 8;
#pragma unroll
            for (int r = 0; r < 8; ++r) {
                const float v = acc[i][j][r] + bv;
                if (outF) outF[(size_t)(mb + r) * N + n] = v;
                else      outB[(size_t)(mb + r) * N + n] = (__bf16)v;
            }
        }
    }
}

// ---------------- fused RMSNorm + rotary, in-place on bf16 [S, C] ----------------
__global__ __launch_bounds__(256)
void qk_norm_rope(__bf16* __restrict__ t, const float* __restrict__ rope,
                  const float* __restrict__ w) {
    const int gw   = (int)((blockIdx.x * 256 + threadIdx.x) >> 5); // wave over S*H
    const int lane = threadIdx.x & 31;
    const int s = gw / Hx;
    const int h = gw % Hx;

    __bf16* row = t + (size_t)s * Cx + h * HDx;
    float x[4];
#pragma unroll
    for (int i = 0; i < 4; ++i) x[i] = (float)row[lane * 4 + i];

    float ss = x[0]*x[0] + x[1]*x[1] + x[2]*x[2] + x[3]*x[3];
#pragma unroll
    for (int off = 1; off < 32; off <<= 1) ss += __shfl_xor(ss, off, 32);
    const float rn = rsqrtf(ss * (1.0f / HDx) + EPSx);

#pragma unroll
    for (int pp = 0; pp < 2; ++pp) {
        const int p = lane * 2 + pp;            // rotary pair index 0..63
        const float x0 = x[pp * 2 + 0] * rn * w[2 * p + 0];
        const float x1 = x[pp * 2 + 1] * rn * w[2 * p + 1];
        const float* rp = rope + ((size_t)s * 64 + p) * 4; // [i][j] 2x2
        row[2 * p + 0] = (__bf16)(rp[0] * x0 + rp[1] * x1);
        row[2 * p + 1] = (__bf16)(rp[2] * x0 + rp[3] * x1);
    }
}

// ---------------- V transpose: [S, C] -> [H][HD][S] (bf16) ----------------
__global__ __launch_bounds__(256)
void v_transpose(const __bf16* __restrict__ v, __bf16* __restrict__ vt) {
    __shared__ __bf16 tile[32][33];
    const int s0 = blockIdx.x * 32;
    const int d0 = blockIdx.y * 32;
    const int h  = blockIdx.z;
    const int r  = threadIdx.x >> 3;
    const int c4 = (threadIdx.x & 7) * 4;
#pragma unroll
    for (int i = 0; i < 4; ++i)
        tile[r][c4 + i] = v[(size_t)(s0 + r) * Cx + h * HDx + d0 + c4 + i];
    __syncthreads();
#pragma unroll
    for (int i = 0; i < 4; ++i)
        vt[((size_t)h * HDx + d0 + r) * Sx + s0 + c4 + i] = tile[c4 + i][r];
}

// ---------------- flash attention: Q[S,C], K[S,C], Vt[H][HD][S] -> O[S,C] (all bf16) ----------------
#define QB 128
#define KBLK 64

__global__ __launch_bounds__(256)
void flash_attn(const __bf16* __restrict__ Q, const __bf16* __restrict__ Kk,
                const __bf16* __restrict__ Vt, __bf16* __restrict__ O) {
    __shared__ alignas(16) __bf16 Ks[2][KBLK][HDx];   // 32 KB: [key][d]
    __shared__ alignas(16) __bf16 Vs[2][HDx][KBLK];   // 32 KB: [d][key]
    __shared__ alignas(16) __bf16 Ps[8][16][KBLK];    // 16 KB: per-wave P staging

    const int h  = blockIdx.y;
    const int q0 = blockIdx.x * QB;
    const int tid  = threadIdx.x;
    const int wave = tid >> 5;
    const int lane = tid & 31;
    const int hi   = lane >> 4;
    const int lr   = lane & 15;
    const float scale = 0.08838834764831845f; // 1/sqrt(128)

    // Q fragments held in registers: wave owns rows q0+wave*16 .. +15
    const __bf16* qp = Q + (size_t)(q0 + wave * 16 + lr) * Cx + h * HDx;
    bf16x16 qf[4];
#pragma unroll
    for (int dc = 0; dc < 4; ++dc)
        qf[dc] = frag_from(qp + dc * 32 + hi * 8, qp + dc * 32 + 16 + hi * 8);

    const f32x8 zero8 = {0.f,0.f,0.f,0.f,0.f,0.f,0.f,0.f};
    f32x8 o[8];
#pragma unroll
    for (int i = 0; i < 8; ++i) o[i] = zero8;
    float mrow[8], lrow[8];
#pragma unroll
    for (int r = 0; r < 8; ++r) { mrow[r] = -3.0e38f; lrow[r] = 0.f; }

    // async staging of K block [64 keys][128 d] and V^T block [128 d][64 keys]
    const int kr = tid >> 2, kc = (tid & 3) * 32;
    const int vr = tid >> 1, vc = (tid & 1) * 32;
    const __bf16* kgbase = Kk + (size_t)kr * Cx + h * HDx + kc;
    const __bf16* vgbase = Vt + ((size_t)h * HDx + vr) * Sx + vc;

    auto stage = [&](int buf, int t0) {
#pragma unroll
        for (int u = 0; u < 4; ++u) {
            async_load_b128(kgbase + (size_t)t0 * Cx + u * 8, &Ks[buf][kr][kc + u * 8]);
            async_load_b128(vgbase + t0 + u * 8,              &Vs[buf][vr][vc + u * 8]);
        }
    };

    const int NT = Sx / KBLK;
    stage(0, 0);
    for (int t = 0; t < NT; ++t) {
        const int cur = t & 1;
        if (t + 1 < NT) {
            stage(cur ^ 1, (t + 1) * KBLK);
            WAIT_ASYNCCNT(8);            // block t resident; block t+1 in flight
        } else {
            WAIT_ASYNCCNT(0);
        }
        __syncthreads();

        // scores: 4 tiles of 16 keys, K-dim = HD in 4 steps of 32
        f32x8 s[4];
#pragma unroll
        for (int kt = 0; kt < 4; ++kt) {
            s[kt] = zero8;
#pragma unroll
            for (int dc = 0; dc < 4; ++dc) {
                bf16x16 kf = frag_from(&Ks[cur][kt * 16 + lr][dc * 32 + hi * 8],
                                       &Ks[cur][kt * 16 + lr][dc * 32 + 16 + hi * 8]);
                s[kt] = wmma_bf16(qf[dc], kf, s[kt]);
            }
        }

        // online softmax; lane element r is row (r + 8*hi), col lr of each tile
        float mnew[8], alpha[8];
#pragma unroll
        for (int r = 0; r < 8; ++r) {
            float mx = fmaxf(fmaxf(s[0][r], s[1][r]), fmaxf(s[2][r], s[3][r]));
#pragma unroll
            for (int off = 1; off < 16; off <<= 1) mx = fmaxf(mx, __shfl_xor(mx, off, 32));
            mx *= scale;
            mnew[r]  = fmaxf(mrow[r], mx);
            alpha[r] = __expf(mrow[r] - mnew[r]);
            mrow[r]  = mnew[r];
        }
#pragma unroll
        for (int r = 0; r < 8; ++r) {
            float rs = 0.f;
            const int prow = r + 8 * hi;
#pragma unroll
            for (int kt = 0; kt < 4; ++kt) {
                const float p = __expf(scale * s[kt][r] - mnew[r]);
                rs += p;
                Ps[wave][prow][kt * 16 + lr] = (__bf16)p;
            }
#pragma unroll
            for (int off = 1; off < 16; off <<= 1) rs += __shfl_xor(rs, off, 32);
            lrow[r] = lrow[r] * alpha[r] + rs;
        }
#pragma unroll
        for (int nt = 0; nt < 8; ++nt)
#pragma unroll
            for (int r = 0; r < 8; ++r) o[nt][r] *= alpha[r];

        // wave-private LDS RAW: wait on DS counter (P staging is per-wave)
        asm volatile("s_wait_dscnt 0" ::: "memory");
        __builtin_amdgcn_wave_barrier();

        // O += P * V : P[16 x 64] from LDS, V^T fragments contiguous in key dim
#pragma unroll
        for (int ts = 0; ts < 2; ++ts) {
            bf16x16 pf = frag_from(&Ps[wave][lr][ts * 32 + hi * 8],
                                   &Ps[wave][lr][ts * 32 + 16 + hi * 8]);
#pragma unroll
            for (int nt = 0; nt < 8; ++nt) {
                bf16x16 vf = frag_from(&Vs[cur][nt * 16 + lr][ts * 32 + hi * 8],
                                       &Vs[cur][nt * 16 + lr][ts * 32 + 16 + hi * 8]);
                o[nt] = wmma_bf16(pf, vf, o[nt]);
            }
        }
        __syncthreads();   // before buffer 'cur' is overwritten at t+2's stage
    }

    // normalize and write attn output [S, C] bf16
#pragma unroll
    for (int r = 0; r < 8; ++r) {
        const float inv = 1.0f / lrow[r];
        const int m = q0 + wave * 16 + 8 * hi + r;
#pragma unroll
        for (int nt = 0; nt < 8; ++nt)
            O[(size_t)m * Cx + h * HDx + nt * 16 + lr] = (__bf16)(o[nt][r] * inv);
    }
}

// ---------------- host-side orchestration ----------------
extern "C" void kernel_launch(void* const* d_in, const int* in_sizes, int n_in,
                              void* d_out, int out_size, void* d_ws, size_t ws_size,
                              hipStream_t stream) {
    const float* x    = (const float*)d_in[0];
    const float* rope = (const float*)d_in[1];
    const float* wq   = (const float*)d_in[2];
    const float* bq   = (const float*)d_in[3];
    const float* wk   = (const float*)d_in[4];
    const float* bk   = (const float*)d_in[5];
    const float* wv   = (const float*)d_in[6];
    const float* bv   = (const float*)d_in[7];
    const float* wo   = (const float*)d_in[8];
    const float* bo   = (const float*)d_in[9];
    const float* qn_w = (const float*)d_in[10];
    const float* kn_w = (const float*)d_in[11];
    float* out = (float*)d_out;

    const size_t SC = (size_t)Sx * Cx;       // 6.29M elems
    const size_t CC = (size_t)Cx * Cx;       // 9.44M elems

    char* p = (char*)d_ws;
    auto alloc = [&](size_t bytes) { char* r = p; p += (bytes + 255) & ~(size_t)255; return r; };
    __bf16* xb  = (__bf16*)alloc(SC * 2);
    __bf16* wqb = (__bf16*)alloc(CC * 2);
    __bf16* wkb = (__bf16*)alloc(CC * 2);
    __bf16* wvb = (__bf16*)alloc(CC * 2);
    __bf16* wob = (__bf16*)alloc(CC * 2);
    __bf16* qb  = (__bf16*)alloc(SC * 2);
    __bf16* kb  = (__bf16*)alloc(SC * 2);
    __bf16* vb  = (__bf16*)alloc(SC * 2);
    __bf16* vtb = (__bf16*)alloc(SC * 2);
    __bf16* ab  = (__bf16*)alloc(SC * 2);

    // 1) fp32 -> bf16
    cvt_f32_bf16<<<(int)(SC / 1024), 256, 0, stream>>>(x,  xb,  SC);
    cvt_f32_bf16<<<(int)(CC / 1024), 256, 0, stream>>>(wq, wqb, CC);
    cvt_f32_bf16<<<(int)(CC / 1024), 256, 0, stream>>>(wk, wkb, CC);
    cvt_f32_bf16<<<(int)(CC / 1024), 256, 0, stream>>>(wv, wvb, CC);
    cvt_f32_bf16<<<(int)(CC / 1024), 256, 0, stream>>>(wo, wob, CC);

    // 2) QKV projections (bf16 out)
    dim3 ggrid(Sx / GTM, Cx / GTN);
    gemm_nt_bias<<<ggrid, 256, 0, stream>>>(xb, wqb, bq, nullptr, qb, Sx, Cx, Cx);
    gemm_nt_bias<<<ggrid, 256, 0, stream>>>(xb, wkb, bk, nullptr, kb, Sx, Cx, Cx);
    gemm_nt_bias<<<ggrid, 256, 0, stream>>>(xb, wvb, bv, nullptr, vb, Sx, Cx, Cx);

    // 3) RMSNorm + rotary on q, k (in place); transpose v
    qk_norm_rope<<<(Sx * Hx) / 8, 256, 0, stream>>>(qb, rope, qn_w);
    qk_norm_rope<<<(Sx * Hx) / 8, 256, 0, stream>>>(kb, rope, kn_w);
    v_transpose<<<dim3(Sx / 32, HDx / 32, Hx), 256, 0, stream>>>(vb, vtb);

    // 4) flash attention
    flash_attn<<<dim3(Sx / QB, Hx), 256, 0, stream>>>(qb, kb, vtb, ab);

    // 5) output projection (fp32 out)
    gemm_nt_bias<<<ggrid, 256, 0, stream>>>(ab, wob, bo, out, nullptr, Sx, Cx, Cx);
}
